// Full_Model_34531537059859
// MI455X (gfx1250) — matrix operand order repaired and checked by reading, not compiled, not run on hardware
//
#include <hip/hip_runtime.h>
#include <hip/hip_bf16.h>
#include <math.h>

// Model dims
#define VSZ 50257
#define DSZ 2048
#define HN  16
#define HDD 128
#define FSZ 8192
#define LNUM 2
#define SSZ 2048
#define LSTR 40   // LDS row stride in bf16 elems (80B = 16B multiple, padded vs 32)

typedef float    f32x4   __attribute__((ext_vector_type(4)));
typedef unsigned u32x4   __attribute__((ext_vector_type(4)));
typedef float    f32x8   __attribute__((ext_vector_type(8)));
typedef __bf16   bf16x2  __attribute__((ext_vector_type(2)));
typedef __bf16   bf16x16 __attribute__((ext_vector_type(16)));

union FragU { bf16x16 v; u32x4 q[2]; };

// Native hardware f32 -> bf16 conversion (v_cvt_pk_bf16_f32 on gfx1250)
__device__ __forceinline__ unsigned short f2bf(float f) {
  __bf16 h = (__bf16)f;
  return __builtin_bit_cast(unsigned short, h);
}
__device__ __forceinline__ unsigned pk2(float a, float b) {
  bf16x2 v = { (__bf16)a, (__bf16)b };
  return __builtin_bit_cast(unsigned, v);
}
__device__ __forceinline__ float bf2f(unsigned short s) {
  return __builtin_bit_cast(float, (unsigned)s << 16);
}

__device__ __forceinline__ f32x8 wmma_bf16(bf16x16 a, bf16x16 b, f32x8 c) {
  return __builtin_amdgcn_wmma_f32_16x16x32_bf16(false, a, false, b, (short)0, c,
                                                 false, false);
}

// ---------------------------------------------------------------------------
// GEMM: C[M,N] = A[M,K] @ B + (RES ? Res : 0)
//   A is bf16 [M,K] row-major -> A fragments load straight from global
//   BT==0: B fp32 [K,N];  BT==1: B fp32 [N,K] (C = A @ B^T, ragged N ok)
//   OUTBF: write C as bf16 instead of fp32
// Tile 128x128, BK=32, 256 threads = 8 waves (4m x 2n), each wave 32x64:
// 8 WMMAs per k-step vs 2 A-frag global loads + 4 B-frag LDS reads.
// Only B goes through LDS (weights, converted fp32->bf16 once per stage).
// ---------------------------------------------------------------------------
template<int BT, int RES, int OUTBF>
__global__ __launch_bounds__(256)
void gemm_bf16_kernel(const unsigned short* __restrict__ A,
                      const float* __restrict__ B,
                      const float* __restrict__ Res, void* __restrict__ Cv,
                      int M, int N, int K)
{
  __shared__ alignas(16) unsigned short Bs[128 * LSTR];

  const int tid  = threadIdx.x;
  const int lane = tid & 31, wid = tid >> 5;
  const int ln16 = lane & 15, lh = lane >> 4;
  const int wm = wid & 3, wn = wid >> 2;
  const int m0 = blockIdx.y * 128, n0 = blockIdx.x * 128;

  const f32x8 z = {0.f,0.f,0.f,0.f,0.f,0.f,0.f,0.f};
  f32x8 acc[2][4];
  #pragma unroll
  for (int i = 0; i < 2; ++i)
    #pragma unroll
    for (int j = 0; j < 4; ++j) acc[i][j] = z;

  // per-lane A row pointers (A-fragment layout: row = ln16, k at 8*lh / 16+8*lh)
  const unsigned short* arow0 = A + (size_t)(m0 + wm*32 + ln16) * K;
  const unsigned short* arow1 = arow0 + (size_t)16 * K;

  for (int k0 = 0; k0 < K; k0 += 32) {
    // ---- stage B into LDS, k-contiguous per output column: Bs[n][k] ----
    if (BT) {
      // 128 rows x 32 k, 16 elems per thread
      int nn = tid >> 1, c0 = (tid & 1) * 16;
      unsigned* dst = (unsigned*)&Bs[nn * LSTR + c0];
      if (n0 + nn < N) {
        const float* src = B + (size_t)(n0 + nn) * K + k0 + c0;
        #pragma unroll
        for (int j = 0; j < 4; ++j) {
          f32x4 f = ((const f32x4*)src)[j];
          dst[2*j]   = pk2(f.x, f.y);
          dst[2*j+1] = pk2(f.z, f.w);
        }
      } else {
        #pragma unroll
        for (int j = 0; j < 8; ++j) dst[j] = 0u;
      }
    } else {
      // B row-major [K,N]: transpose while staging; 16 consecutive n per thread
      int kk = tid >> 3, nn = (tid & 7) * 16;
      const float* src = B + (size_t)(k0 + kk) * N + n0 + nn;
      if (k0 + 32 < K) __builtin_prefetch(src + (size_t)32 * N, 0, 0);
      #pragma unroll
      for (int j4 = 0; j4 < 4; ++j4) {
        f32x4 f = ((const f32x4*)src)[j4];
        Bs[(nn + 4*j4 + 0) * LSTR + kk] = f2bf(f.x);
        Bs[(nn + 4*j4 + 1) * LSTR + kk] = f2bf(f.y);
        Bs[(nn + 4*j4 + 2) * LSTR + kk] = f2bf(f.z);
        Bs[(nn + 4*j4 + 3) * LSTR + kk] = f2bf(f.w);
      }
    }
    __syncthreads();

    // ---- A fragments straight from global (bf16, no conversion) ----
    FragU af[2], bfr[4];
    af[0].q[0] = *(const u32x4*)(arow0 + k0 + 8*lh);
    af[0].q[1] = *(const u32x4*)(arow0 + k0 + 16 + 8*lh);
    af[1].q[0] = *(const u32x4*)(arow1 + k0 + 8*lh);
    af[1].q[1] = *(const u32x4*)(arow1 + k0 + 16 + 8*lh);
    #pragma unroll
    for (int fn = 0; fn < 4; ++fn) {
      const unsigned short* pb = &Bs[(wn*64 + fn*16 + ln16) * LSTR + 8*lh];
      bfr[fn].q[0] = *(const u32x4*)(pb);
      bfr[fn].q[1] = *(const u32x4*)(pb + 16);
    }
    #pragma unroll
    for (int fm = 0; fm < 2; ++fm)
      #pragma unroll
      for (int fn = 0; fn < 4; ++fn)
        acc[fm][fn] = wmma_bf16(af[fm].v, bfr[fn].v, acc[fm][fn]);
    __syncthreads();
  }

  // ---- epilogue: C layout m = r + 8*lh, n = ln16 ----
  #pragma unroll
  for (int fm = 0; fm < 2; ++fm) {
    int rowb = m0 + wm*32 + fm*16 + 8*lh;
    #pragma unroll
    for (int fn = 0; fn < 4; ++fn) {
      int col = n0 + wn*64 + fn*16 + ln16;
      if (col < N) {
        #pragma unroll
        for (int r = 0; r < 8; ++r) {
          float vv = acc[fm][fn][r];
          size_t idx = (size_t)(rowb + r) * N + col;
          if (RES) vv += Res[idx];
          if (OUTBF) ((unsigned short*)Cv)[idx] = f2bf(vv);
          else       ((float*)Cv)[idx] = vv;
        }
      }
    }
  }
}

// ---------------------------------------------------------------------------
// Flash attention: one wave per 16-row query block per head.
// Q,K,V bf16 [S,D]; O bf16 [S,D]. Head h = columns [h*128, h*128+128).
// Zero conversions in the inner loop except P (exp output).
// ---------------------------------------------------------------------------
__global__ __launch_bounds__(32)
void flash_attn_kernel(const unsigned short* __restrict__ Q,
                       const unsigned short* __restrict__ K,
                       const unsigned short* __restrict__ Vv,
                       unsigned short* __restrict__ O)
{
  __shared__ alignas(16) unsigned short Vs[HDD * LSTR];  // Vs[dim][key]
  __shared__ alignas(16) unsigned short Ps[16 * LSTR];   // Ps[m][key]
  const int lane = threadIdx.x;
  const int ln16 = lane & 15, lh = lane >> 4;
  const int qb = blockIdx.x, hd = blockIdx.y;
  const int q0 = qb * 16;
  const float scale = 0.088388347648318447f;  // 1/sqrt(128)

  // Q fragments (A layout) straight from global bf16
  FragU qf[4];
  {
    const unsigned short* qp = Q + (size_t)(q0 + ln16) * DSZ + hd * HDD;
    #pragma unroll
    for (int c = 0; c < 4; ++c) {
      qf[c].q[0] = *(const u32x4*)(qp + c*32 + 8*lh);
      qf[c].q[1] = *(const u32x4*)(qp + c*32 + 16 + 8*lh);
    }
  }

  const f32x8 z = {0.f,0.f,0.f,0.f,0.f,0.f,0.f,0.f};
  f32x8 out[8];
  #pragma unroll
  for (int ng = 0; ng < 8; ++ng) out[ng] = z;
  float Mrow[8], Lrow[8];
  #pragma unroll
  for (int r = 0; r < 8; ++r) { Mrow[r] = -1e30f; Lrow[r] = 0.f; }

  const int nk = q0 + 16;  // causal: keys [0, q0+16)
  for (int kbase = 0; kbase < nk; kbase += 32) {
    // ---- scores: two 16x16 tiles over 32 keys; K frags direct from global ----
    f32x8 s[2]; s[0] = z; s[1] = z;
    #pragma unroll
    for (int t = 0; t < 2; ++t) {
      const unsigned short* kp = K + (size_t)(kbase + t*16 + ln16) * DSZ + hd * HDD;
      #pragma unroll
      for (int c = 0; c < 4; ++c) {
        FragU kf;  // B layout: k at 8*lh / 16+8*lh
        kf.q[0] = *(const u32x4*)(kp + c*32 + 8*lh);
        kf.q[1] = *(const u32x4*)(kp + c*32 + 16 + 8*lh);
        s[t] = wmma_bf16(qf[c].v, kf.v, s[t]);
      }
    }
    // ---- stage V transposed into LDS (bf16 copy, no cvt): Vs[dim][key] ----
    {
      const unsigned short* vp = Vv + (size_t)(kbase + lane) * DSZ + hd * HDD;
      #pragma unroll
      for (int c = 0; c < 8; ++c) {
        u32x4 d = *(const u32x4*)(vp + c*8);   // dims c*8 .. c*8+7
        unsigned xs[4] = {d.x, d.y, d.z, d.w};
        #pragma unroll
        for (int w = 0; w < 4; ++w) {
          Vs[(c*8 + 2*w    )*LSTR + lane] = (unsigned short)(xs[w] & 0xffffu);
          Vs[(c*8 + 2*w + 1)*LSTR + lane] = (unsigned short)(xs[w] >> 16);
        }
      }
    }
    // ---- online softmax (rows r+8*lh, one column per lane group) ----
    #pragma unroll
    for (int r = 0; r < 8; ++r) {
      int qrow = q0 + r + 8*lh;
      float sc0 = s[0][r] * scale;
      float sc1 = s[1][r] * scale;
      if (kbase + ln16 > qrow)      sc0 = -1e30f;
      if (kbase + 16 + ln16 > qrow) sc1 = -1e30f;
      float mx = fmaxf(sc0, sc1);
      #pragma unroll
      for (int off = 1; off < 16; off <<= 1) mx = fmaxf(mx, __shfl_xor(mx, off, 32));
      float mnew = fmaxf(Mrow[r], mx);
      float corr = __expf(Mrow[r] - mnew);
      float p0 = __expf(sc0 - mnew);
      float p1 = __expf(sc1 - mnew);
      float rs = p0 + p1;
      #pragma unroll
      for (int off = 1; off < 16; off <<= 1) rs += __shfl_xor(rs, off, 32);
      Lrow[r] = Lrow[r] * corr + rs;
      Mrow[r] = mnew;
      #pragma unroll
      for (int ng = 0; ng < 8; ++ng) out[ng][r] *= corr;
      int m = r + 8*lh;
      Ps[m*LSTR + ln16]      = f2bf(p0);
      Ps[m*LSTR + 16 + ln16] = f2bf(p1);
    }
    // ---- P @ V (A from Ps, B from Vs) ----
    FragU pf;
    const unsigned short* pp = &Ps[ln16 * LSTR];
    pf.q[0] = *(const u32x4*)(pp + 8*lh);
    pf.q[1] = *(const u32x4*)(pp + 16 + 8*lh);
    #pragma unroll
    for (int ng = 0; ng < 8; ++ng) {
      FragU vf;
      const unsigned short* vb = &Vs[(ng*16 + ln16) * LSTR + 8*lh];
      vf.q[0] = *(const u32x4*)(vb);
      vf.q[1] = *(const u32x4*)(vb + 16);
      out[ng] = wmma_bf16(pf.v, vf.v, out[ng]);
    }
  }

  #pragma unroll
  for (int ng = 0; ng < 8; ++ng) {
    #pragma unroll
    for (int r = 0; r < 8; ++r) {
      int row = q0 + r + 8*lh;
      O[(size_t)row * DSZ + hd*HDD + ng*16 + ln16] = f2bf(out[ng][r] / Lrow[r]);
    }
  }
}

// ---------------------------------------------------------------------------
// Elementwise / reduction kernels
// ---------------------------------------------------------------------------
__global__ void embed_kernel(const int* __restrict__ ids,
                             const float* __restrict__ table,
                             float* __restrict__ H)
{
  int idx = blockIdx.x * 256 + threadIdx.x;
  int s = idx >> 11;          // / 2048
  int d = idx & 2047;
  H[idx] = table[(size_t)ids[s] * DSZ + d];
}

// RMSNorm: fp32 in (residual stream), bf16 out (GEMM A operand)
__global__ __launch_bounds__(256)
void rmsnorm_kernel(const float* __restrict__ X, const float* __restrict__ W,
                    unsigned short* __restrict__ Y)
{
  __shared__ float red[256];
  int s = blockIdx.x, t = threadIdx.x;
  const float* x = X + (size_t)s * DSZ;
  float ss = 0.f;
  for (int i = t; i < DSZ; i += 256) { float v = x[i]; ss += v * v; }
  red[t] = ss; __syncthreads();
  for (int w = 128; w > 0; w >>= 1) {
    if (t < w) red[t] += red[t + w];
    __syncthreads();
  }
  float sc = rsqrtf(red[0] / (float)DSZ + 1e-6f);
  for (int i = t; i < DSZ; i += 256)
    Y[(size_t)s * DSZ + i] = f2bf(x[i] * sc * W[i]);
}

// RoPE in place on bf16 q,k
__global__ void rope_kernel(unsigned short* __restrict__ Q,
                            unsigned short* __restrict__ K)
{
  int idx = blockIdx.x * 256 + threadIdx.x;   // S*H*64 threads
  int j = idx & 63;
  int h = (idx >> 6) & (HN - 1);
  int s = idx >> 10;
  float inv = __expf(-logf(10000.0f) * (float)(2 * j) * (1.0f / 128.0f));
  float ang = (float)s * inv;
  float c = cosf(ang), si = sinf(ang);
  size_t base = (size_t)s * DSZ + h * HDD + j;
  float q1 = bf2f(Q[base]), q2 = bf2f(Q[base + 64]);
  Q[base]      = f2bf(q1 * c - q2 * si);
  Q[base + 64] = f2bf(q2 * c + q1 * si);
  float k1 = bf2f(K[base]), k2 = bf2f(K[base + 64]);
  K[base]      = f2bf(k1 * c - k2 * si);
  K[base + 64] = f2bf(k2 * c + k1 * si);
}

// silu(g)*u : fp32 in, bf16 out (A operand of the down projection)
__global__ void silu_mul_kernel(const float* __restrict__ G,
                                const float* __restrict__ U,
                                unsigned short* __restrict__ GA)
{
  int idx = blockIdx.x * 256 + threadIdx.x;
  float g = G[idx];
  GA[idx] = f2bf((g / (1.0f + __expf(-g))) * U[idx]);
}

__global__ __launch_bounds__(256)
void row_nll_kernel(const float* __restrict__ logits, const int* __restrict__ labels,
                    float* __restrict__ nll)
{
  __shared__ float red[256];
  int s = blockIdx.x, t = threadIdx.x;           // s in [0, S-1)
  const float* row = logits + (size_t)s * VSZ;
  float mx = -1e30f;
  for (int i = t; i < VSZ; i += 256) mx = fmaxf(mx, row[i]);
  red[t] = mx; __syncthreads();
  for (int w = 128; w > 0; w >>= 1) {
    if (t < w) red[t] = fmaxf(red[t], red[t + w]);
    __syncthreads();
  }
  mx = red[0]; __syncthreads();
  float se = 0.f;
  for (int i = t; i < VSZ; i += 256) se += __expf(row[i] - mx);
  red[t] = se; __syncthreads();
  for (int w = 128; w > 0; w >>= 1) {
    if (t < w) red[t] += red[t + w];
    __syncthreads();
  }
  if (t == 0) {
    int lab = labels[s + 1];
    nll[s] = logf(red[0]) + mx - row[lab];
  }
}

__global__ __launch_bounds__(256)
void loss_finalize_kernel(const float* __restrict__ nll, float* __restrict__ loss)
{
  __shared__ float red[256];
  int t = threadIdx.x;
  float acc = 0.f;
  for (int i = t; i < SSZ - 1; i += 256) acc += nll[i];
  red[t] = acc; __syncthreads();
  for (int w = 128; w > 0; w >>= 1) {
    if (t < w) red[t] += red[t + w];
    __syncthreads();
  }
  if (t == 0) *loss = red[0] / (float)(SSZ - 1);
}

// ---------------------------------------------------------------------------
// Host orchestration
// ---------------------------------------------------------------------------
extern "C" void kernel_launch(void* const* d_in, const int* in_sizes, int n_in,
                              void* d_out, int out_size, void* d_ws, size_t ws_size,
                              hipStream_t stream)
{
  (void)in_sizes; (void)n_in; (void)out_size; (void)ws_size;
  const int*   input_ids = (const int*)d_in[0];
  const int*   label_ids = (const int*)d_in[1];
  const float* embed_tab = (const float*)d_in[2];
  const float* Wq = (const float*)d_in[3];
  const float* Wk = (const float*)d_in[4];
  const float* Wv = (const float*)d_in[5];
  const float* Wo = (const float*)d_in[6];
  const float* Wg = (const float*)d_in[7];
  const float* Wu = (const float*)d_in[8];
  const float* Wd = (const float*)d_in[9];
  const float* ln1 = (const float*)d_in[10];
  const float* ln2 = (const float*)d_in[11];
  const float* norm_w = (const float*)d_in[12];
  const float* lm_head = (const float*)d_in[13];

  float* loss   = (float*)d_out;
  float* logits = (float*)d_out + 1;

  const size_t SD = (size_t)SSZ * DSZ;
  const size_t SF = (size_t)SSZ * FSZ;
  // fp32 region
  float* ws = (float*)d_ws;
  float* h   = ws;            // residual stream
  float* g   = h + SD;        // mlp gate (pre-activation)
  float* u   = g + SF;        // mlp up
  float* nll = u + SF;        // (S-1) floats
  // bf16 region (starts 16B-aligned: (SD + 2*SF + 2048)*4 bytes in)
  unsigned short* bws = (unsigned short*)(nll + 2048);
  unsigned short* xnb  = bws;         // normed activations
  unsigned short* qb   = xnb  + SD;
  unsigned short* kb   = qb   + SD;
  unsigned short* vb   = kb   + SD;
  unsigned short* ob   = vb   + SD;   // attention output
  unsigned short* gact = ob   + SD;   // silu(g)*u

  dim3 blk256(256), blk32(32);
  dim3 gDD(DSZ / 128, SSZ / 128);                // N=2048 GEMMs
  dim3 gDF(FSZ / 128, SSZ / 128);                // N=8192 GEMMs
  dim3 gV((VSZ + 127) / 128, SSZ / 128);         // lm_head GEMM
  dim3 gAttn(SSZ / 16, HN);

  embed_kernel<<<(int)(SD / 256), blk256, 0, stream>>>(input_ids, embed_tab, h);

  for (int l = 0; l < LNUM; ++l) {
    const float* wq = Wq + (size_t)l * DSZ * DSZ;
    const float* wk = Wk + (size_t)l * DSZ * DSZ;
    const float* wv = Wv + (size_t)l * DSZ * DSZ;
    const float* wo = Wo + (size_t)l * DSZ * DSZ;
    const float* wg = Wg + (size_t)l * DSZ * FSZ;
    const float* wu = Wu + (size_t)l * DSZ * FSZ;
    const float* wd = Wd + (size_t)l * FSZ * DSZ;

    rmsnorm_kernel<<<SSZ, blk256, 0, stream>>>(h, ln1 + (size_t)l * DSZ, xnb);
    gemm_bf16_kernel<0,0,1><<<gDD, blk256, 0, stream>>>(xnb, wq, nullptr, qb, SSZ, DSZ, DSZ);
    gemm_bf16_kernel<0,0,1><<<gDD, blk256, 0, stream>>>(xnb, wk, nullptr, kb, SSZ, DSZ, DSZ);
    gemm_bf16_kernel<0,0,1><<<gDD, blk256, 0, stream>>>(xnb, wv, nullptr, vb, SSZ, DSZ, DSZ);
    rope_kernel<<<(SSZ * HN * 64) / 256, blk256, 0, stream>>>(qb, kb);
    flash_attn_kernel<<<gAttn, blk32, 0, stream>>>(qb, kb, vb, ob);
    gemm_bf16_kernel<0,1,0><<<gDD, blk256, 0, stream>>>(ob, wo, h, h, SSZ, DSZ, DSZ);

    rmsnorm_kernel<<<SSZ, blk256, 0, stream>>>(h, ln2 + (size_t)l * DSZ, xnb);
    gemm_bf16_kernel<0,0,0><<<gDF, blk256, 0, stream>>>(xnb, wg, nullptr, g, SSZ, FSZ, DSZ);
    gemm_bf16_kernel<0,0,0><<<gDF, blk256, 0, stream>>>(xnb, wu, nullptr, u, SSZ, FSZ, DSZ);
    silu_mul_kernel<<<(int)(SF / 256), blk256, 0, stream>>>(g, u, gact);
    gemm_bf16_kernel<0,1,0><<<gDD, blk256, 0, stream>>>(gact, wd, h, h, SSZ, DSZ, FSZ);
  }

  rmsnorm_kernel<<<SSZ, blk256, 0, stream>>>(h, norm_w, xnb);
  gemm_bf16_kernel<1,0,0><<<gV, blk256, 0, stream>>>(xnb, lm_head, nullptr, logits,
                                                     SSZ, VSZ, DSZ);

  row_nll_kernel<<<SSZ - 1, blk256, 0, stream>>>(logits, label_ids, nll);
  loss_finalize_kernel<<<1, blk256, 0, stream>>>(nll, loss);
}
